// BigSSMBlock_24739011625089
// MI455X (gfx1250) — matrix operand order repaired and checked by reading, not compiled, not run on hardware
//
#include <hip/hip_runtime.h>
#include <hip/hip_bf16.h>
#include <math.h>
#include <stdint.h>

#define D_MODEL 1024
#define SEQ     1024
#define DSTATE  64
#define EPS_F   1e-6f

typedef __attribute__((ext_vector_type(16))) __bf16       v16bf;
typedef __attribute__((ext_vector_type(8)))  float        v8f;
typedef __attribute__((ext_vector_type(4)))  unsigned int v4u;
typedef __attribute__((ext_vector_type(4)))  unsigned int u32x4;
typedef __attribute__((ext_vector_type(8)))  int          i32x8;
typedef __attribute__((ext_vector_type(4)))  int          i32x4;

#if defined(__has_builtin)
#if __has_builtin(__builtin_amdgcn_tensor_load_to_lds)
#define HAS_TDM 1
#endif
#endif

union Frag { v4u u[2]; v16bf v; };

// ---- helpers -------------------------------------------------------------

static __device__ __forceinline__ unsigned short f2bf(float f) {
  // round-to-nearest-even f32 -> bf16 bit pattern
  unsigned int u = __float_as_uint(f);
  unsigned int r = (u + 0x7FFFu + ((u >> 16) & 1u)) >> 16;
  return (unsigned short)r;
}

// ---- elementwise f32 -> bf16 convert ------------------------------------

__global__ void cvt_f32_bf16(const float* __restrict__ src,
                             unsigned short* __restrict__ dst, int n) {
  int i = blockIdx.x * blockDim.x + threadIdx.x;
  if (i < n) dst[i] = f2bf(src[i]);
}

// ---- RMSNorm (one block per token row) + bf16 convert --------------------

__global__ void __launch_bounds__(256)
rmsnorm_bf16(const float* __restrict__ x, const float* __restrict__ w,
             unsigned short* __restrict__ h) {
  __shared__ float red[256];
  int row = blockIdx.x;
  int tid = threadIdx.x;
  const float* xr = x + (size_t)row * D_MODEL;
  float s = 0.f;
  for (int i = tid; i < D_MODEL; i += 256) { float v = xr[i]; s += v * v; }
  red[tid] = s;
  __syncthreads();
  for (int o = 128; o > 0; o >>= 1) {
    if (tid < o) red[tid] += red[tid + o];
    __syncthreads();
  }
  float inv = rsqrtf(red[0] / (float)D_MODEL + EPS_F);
  for (int i = tid; i < D_MODEL; i += 256)
    h[(size_t)row * D_MODEL + i] = f2bf(xr[i] * inv * w[i]);
}

// ---- bf16 WMMA GEMM: C[M,N] = A[M,K] * W[N,K]^T (f32 accumulate) ---------
// Block = 8 waves = 256 (M) x 64 (N) output. Each wave: 2 M-tiles x 4 N-tiles
// (8 accumulators). The 64x32 B panel (shared by all 8 waves) is staged into
// LDS each K-step by the Tensor Data Mover (tensor_load_to_lds + tensorcnt),
// then consumed via ds_load_b128. A fragments stream via global_load_b128.

__global__ void __launch_bounds__(256)
gemm_bf16_wmma(const unsigned short* __restrict__ A,
               const unsigned short* __restrict__ W,
               float* __restrict__ C, int M, int N, int K) {
  __shared__ v4u spanel[256];                    // 64 rows x 32 bf16 = 4 KB
  const int nstrips = N >> 6;
  const int bm   = blockIdx.x / nstrips;
  const int bn   = blockIdx.x - bm * nstrips;
  const int wave = threadIdx.x >> 5;
  const int lane = threadIdx.x & 31;
  const int sel  = lane >> 4;                    // half-select
  const int rm   = lane & 15;
  const int m0   = bm * 256 + wave * 32;         // wave's 32-row strip
  const int n0   = bn * 64;                      // block's 64-col strip
  (void)M;

  // A fragment rows (two 16-row tiles): K = k + sel*8 + [0..8) and +16
  const unsigned short* arow0 = A + (size_t)(m0 + rm) * K + sel * 8;
  const unsigned short* arow1 = arow0 + (size_t)16 * K;

  v8f acc[2][4];
#pragma unroll
  for (int i = 0; i < 2; ++i)
#pragma unroll
    for (int j = 0; j < 4; ++j) acc[i][j] = (v8f){};

  for (int k = 0; k < K; k += 32) {
    // ---- stage B panel: rows n0..n0+63 of W, columns k..k+31 -------------
#ifdef HAS_TDM
    if (threadIdx.x < 32) {                      // one TDM issue per block
      unsigned long long ga =
          (unsigned long long)(uintptr_t)W + ((size_t)n0 * K + k) * 2ull;
      unsigned lds_off = (unsigned)(uintptr_t)&spanel[0];  // low 32b = LDS addr
      u32x4 g0;
      g0[0] = 1u;                                          // count=1, user mode
      g0[1] = lds_off;                                     // lds_addr
      g0[2] = (unsigned)(ga & 0xFFFFFFFFull);              // global_addr[31:0]
      g0[3] = (unsigned)((ga >> 32) & 0x01FFFFFFull)       // global_addr[56:32]
              | (2u << 30);                                // type=2 (image)
      i32x8 g1;
      g1[0] = (int)(1u << 16);                   // data_size=1 (2 bytes)
      g1[1] = (int)(1024u << 16);                // tensor_dim0 = K (low 16)
      g1[2] = (int)(64u << 16);                  // tensor_dim1 = 64
      g1[3] = (int)(32u << 16);                  // tile_dim0 = 32
      g1[4] = 64;                                // tile_dim1 = 64
      g1[5] = 1024;                              // tensor_dim0_stride = K
      g1[6] = 0;
      g1[7] = 0;
      i32x4 z4 = {0, 0, 0, 0};
#if __has_include(<hip/amd_detail/amd_gfx1250_TDM.h>)
      i32x8 z8 = {0, 0, 0, 0, 0, 0, 0, 0};       // 6-arg therock toolchain
      __builtin_amdgcn_tensor_load_to_lds(g0, g1, z4, z4, z8, 0);
#else
      __builtin_amdgcn_tensor_load_to_lds(g0, g1, z4, z4, 0);  // 5-arg ROCm 7.2
#endif
      __builtin_amdgcn_s_wait_tensorcnt(0);
    }
#else
    {                                            // cooperative-copy fallback
      int t   = threadIdx.x;
      int row = t >> 2;
      int cc  = (t & 3) * 8;
      spanel[row * 4 + (cc >> 3)] =
          *(const v4u*)(W + (size_t)(n0 + row) * K + k + cc);
    }
#endif
    __syncthreads();                             // publish panel

    Frag a0, a1, b[4];
    a0.u[0] = *(const v4u*)(arow0 + k);
    a0.u[1] = *(const v4u*)(arow0 + k + 16);
    a1.u[0] = *(const v4u*)(arow1 + k);
    a1.u[1] = *(const v4u*)(arow1 + k + 16);
#pragma unroll
    for (int j = 0; j < 4; ++j) {
      int row = j * 16 + rm;                     // panel row = W row (n-index)
      b[j].u[0] = spanel[row * 4 + sel];         // K = sel*8 + [0..8)
      b[j].u[1] = spanel[row * 4 + sel + 2];     // K = 16 + sel*8 + [0..8)
    }
#pragma unroll
    for (int j = 0; j < 4; ++j) {
      acc[0][j] = __builtin_amdgcn_wmma_f32_16x16x32_bf16(
          false, a0.v, false, b[j].v, (short)0, acc[0][j], false, false);
      acc[1][j] = __builtin_amdgcn_wmma_f32_16x16x32_bf16(
          false, a1.v, false, b[j].v, (short)0, acc[1][j], false, false);
    }
    __syncthreads();                             // drain before next overwrite
  }

  // D layout: VGPR r -> row +sel*8+r, col = tile_n*16 + rm
#pragma unroll
  for (int i = 0; i < 2; ++i)
#pragma unroll
    for (int j = 0; j < 4; ++j) {
      float* crow = C + (size_t)(m0 + i * 16 + sel * 8) * N + n0 + j * 16 + rm;
#pragma unroll
      for (int r = 0; r < 8; ++r) crow[(size_t)r * N] = acc[i][j][r];
    }
}

// ---- causal depthwise conv (k=4) + SiLU ----------------------------------

__global__ void conv_silu(const float* __restrict__ xz,
                          const float* __restrict__ cw,
                          const float* __restrict__ cb,
                          float* __restrict__ xs_f,
                          unsigned short* __restrict__ xs_bf) {
  int idx = blockIdx.x * blockDim.x + threadIdx.x;
  if (idx >= SEQ * D_MODEL) return;
  int t = idx >> 10;
  int d = idx & (D_MODEL - 1);
  float acc = cb[d];
#pragma unroll
  for (int k = 0; k < 4; ++k) {
    int tt = t - 3 + k;
    if (tt >= 0) acc += cw[d * 4 + k] * xz[(size_t)tt * (2 * D_MODEL) + d];
  }
  float s = acc / (1.f + __expf(-acc));          // SiLU
  xs_f[idx]  = s;
  xs_bf[idx] = f2bf(s);
}

// ---- dt = softplus(dt_lin + dt_b), in place ------------------------------

__global__ void dt_softplus(float* __restrict__ dt, const float* __restrict__ dt_b) {
  int idx = blockIdx.x * blockDim.x + threadIdx.x;
  if (idx >= SEQ * D_MODEL) return;
  int d = idx & (D_MODEL - 1);
  float v = dt[idx] + dt_b[d];
  dt[idx] = (v > 20.f) ? v : log1pf(__expf(v));
}

// ---- sequential SSM scan: one thread per channel, 64-state in registers --

__global__ void __launch_bounds__(256)
ssm_scan(const float* __restrict__ xs, const float* __restrict__ dt,
         const float* __restrict__ Bm, const float* __restrict__ Cm,
         const float* __restrict__ log_A, float* __restrict__ y) {
  __shared__ float sB[DSTATE];
  __shared__ float sC[DSTATE];
  int d  = blockIdx.x * 256 + threadIdx.x;
  float Ad = -__expf(log_A[d]);
  float h[DSTATE];
#pragma unroll
  for (int s = 0; s < DSTATE; ++s) h[s] = 0.f;

  for (int t = 0; t < SEQ; ++t) {
    __syncthreads();
    if (threadIdx.x < DSTATE)
      sB[threadIdx.x] = Bm[(size_t)t * DSTATE + threadIdx.x];
    else if (threadIdx.x < 2 * DSTATE)
      sC[threadIdx.x - DSTATE] = Cm[(size_t)t * DSTATE + threadIdx.x - DSTATE];
    __syncthreads();

    float dtv = dt[(size_t)t * D_MODEL + d];
    float la  = fminf(fmaxf(dtv * Ad, -20.f), 20.f);
    float a   = __expf(la);
    float u   = dtv * xs[(size_t)t * D_MODEL + d];
    float acc = 0.f;
#pragma unroll
    for (int s = 0; s < DSTATE; ++s) {
      h[s] = a * h[s] + u * sB[s];
      acc += sC[s] * h[s];
    }
    y[(size_t)t * D_MODEL + d] = acc;
  }
}

// ---- gating: yg = (y + xs*D) * silu(z), to bf16 --------------------------

__global__ void gate_silu(const float* __restrict__ y, const float* __restrict__ xs,
                          const float* __restrict__ xz, const float* __restrict__ Dv,
                          unsigned short* __restrict__ yg) {
  int idx = blockIdx.x * blockDim.x + threadIdx.x;
  if (idx >= SEQ * D_MODEL) return;
  int t = idx >> 10;
  int d = idx & (D_MODEL - 1);
  float z = xz[(size_t)t * (2 * D_MODEL) + D_MODEL + d];
  float g = z / (1.f + __expf(-z));
  yg[idx] = f2bf((y[idx] + xs[idx] * Dv[d]) * g);
}

// ---- residual add --------------------------------------------------------

__global__ void residual_add(const float* __restrict__ x, const float* __restrict__ go,
                             float* __restrict__ out, int n) {
  int i = blockIdx.x * blockDim.x + threadIdx.x;
  if (i < n) out[i] = x[i] + go[i];
}

// ---- launcher ------------------------------------------------------------

extern "C" void kernel_launch(void* const* d_in, const int* in_sizes, int n_in,
                              void* d_out, int out_size, void* d_ws, size_t ws_size,
                              hipStream_t stream) {
  (void)in_sizes; (void)n_in; (void)out_size; (void)ws_size;

  const float* x         = (const float*)d_in[0];
  const float* norm_w    = (const float*)d_in[1];
  const float* in_proj_w = (const float*)d_in[2];   // (2048, 1024)
  const float* conv_w    = (const float*)d_in[3];   // (1024, 1, 4)
  const float* conv_b    = (const float*)d_in[4];
  const float* dt_w      = (const float*)d_in[5];   // (1024, 1024)
  const float* dt_b      = (const float*)d_in[6];
  const float* B_w       = (const float*)d_in[7];   // (64, 1024)
  const float* C_w       = (const float*)d_in[8];   // (64, 1024)
  const float* out_w     = (const float*)d_in[9];   // (1024, 1024)
  const float* log_A     = (const float*)d_in[10];
  const float* Dv        = (const float*)d_in[11];
  float*       out       = (float*)d_out;

  // workspace carve-up (256B aligned slices)
  char* ws = (char*)d_ws;
  size_t off = 0;
  auto alloc = [&](size_t bytes) -> void* {
    void* p = ws + off;
    off = (off + bytes + 255) & ~(size_t)255;
    return p;
  };
  const size_t TD = (size_t)SEQ * D_MODEL;
  unsigned short* h_bf   = (unsigned short*)alloc(TD * 2);                 // rmsnorm out
  unsigned short* win_bf = (unsigned short*)alloc((size_t)2 * TD * 2);     // in_proj_w bf16
  unsigned short* wdt_bf = (unsigned short*)alloc(TD * 2);                 // dt_w bf16
  unsigned short* wb_bf  = (unsigned short*)alloc((size_t)DSTATE * D_MODEL * 2);
  unsigned short* wc_bf  = (unsigned short*)alloc((size_t)DSTATE * D_MODEL * 2);
  unsigned short* wo_bf  = (unsigned short*)alloc(TD * 2);                 // out_w bf16
  float*          xz_f   = (float*)alloc((size_t)2 * TD * 4);              // in_proj output
  float*          xs_f   = (float*)alloc(TD * 4);
  unsigned short* xs_bf  = (unsigned short*)alloc(TD * 2);
  float*          dt_f   = (float*)alloc(TD * 4);
  float*          bm_f   = (float*)alloc((size_t)SEQ * DSTATE * 4);
  float*          cm_f   = (float*)alloc((size_t)SEQ * DSTATE * 4);
  float*          y_f    = (float*)alloc(TD * 4);
  unsigned short* yg_bf  = (unsigned short*)alloc(TD * 2);
  float*          go_f   = (float*)alloc(TD * 4);

  // 1) weight conversion f32 -> bf16
  {
    int n;
    n = 2 * D_MODEL * D_MODEL;
    cvt_f32_bf16<<<(n + 255) / 256, 256, 0, stream>>>(in_proj_w, win_bf, n);
    n = D_MODEL * D_MODEL;
    cvt_f32_bf16<<<(n + 255) / 256, 256, 0, stream>>>(dt_w, wdt_bf, n);
    n = DSTATE * D_MODEL;
    cvt_f32_bf16<<<(n + 255) / 256, 256, 0, stream>>>(B_w, wb_bf, n);
    cvt_f32_bf16<<<(n + 255) / 256, 256, 0, stream>>>(C_w, wc_bf, n);
    n = D_MODEL * D_MODEL;
    cvt_f32_bf16<<<(n + 255) / 256, 256, 0, stream>>>(out_w, wo_bf, n);
  }

  // 2) RMSNorm + bf16
  rmsnorm_bf16<<<SEQ, 256, 0, stream>>>(x, norm_w, h_bf);

  // block tile = 256 (M) x 64 (N); all shapes divide exactly
  auto gemm_blocks = [](int M, int N) { return (M / 256) * (N / 64); };

  // 3) xz = h @ in_proj_w^T   (1024 x 2048)
  gemm_bf16_wmma<<<gemm_blocks(SEQ, 2 * D_MODEL), 256, 0, stream>>>(
      h_bf, win_bf, xz_f, SEQ, 2 * D_MODEL, D_MODEL);

  // 4) xs = silu(causal_dwconv(xz[:, :1024]))
  conv_silu<<<(SEQ * D_MODEL + 255) / 256, 256, 0, stream>>>(
      xz_f, conv_w, conv_b, xs_f, xs_bf);

  // 5) dt = softplus(xs @ dt_w^T + dt_b)
  gemm_bf16_wmma<<<gemm_blocks(SEQ, D_MODEL), 256, 0, stream>>>(
      xs_bf, wdt_bf, dt_f, SEQ, D_MODEL, D_MODEL);
  dt_softplus<<<(SEQ * D_MODEL + 255) / 256, 256, 0, stream>>>(dt_f, dt_b);

  // 6) Bm = xs @ B_w^T ; Cm = xs @ C_w^T   (1024 x 64)
  gemm_bf16_wmma<<<gemm_blocks(SEQ, DSTATE), 256, 0, stream>>>(
      xs_bf, wb_bf, bm_f, SEQ, DSTATE, D_MODEL);
  gemm_bf16_wmma<<<gemm_blocks(SEQ, DSTATE), 256, 0, stream>>>(
      xs_bf, wc_bf, cm_f, SEQ, DSTATE, D_MODEL);

  // 7) sequential SSM scan
  ssm_scan<<<D_MODEL / 256, 256, 0, stream>>>(xs_f, dt_f, bm_f, cm_f, log_A, y_f);

  // 8) gating
  gate_silu<<<(SEQ * D_MODEL + 255) / 256, 256, 0, stream>>>(y_f, xs_f, xz_f, Dv, yg_bf);

  // 9) out GEMM + residual
  gemm_bf16_wmma<<<gemm_blocks(SEQ, D_MODEL), 256, 0, stream>>>(
      yg_bf, wo_bf, go_f, SEQ, D_MODEL, D_MODEL);
  residual_add<<<(SEQ * D_MODEL + 255) / 256, 256, 0, stream>>>(
      x, go_f, out, SEQ * D_MODEL);
}
